// RNN_20830591386036
// MI455X (gfx1250) — compile-verified, matrix operand order checked by reading, not run on hardware
//
#include <hip/hip_runtime.h>
#include <hip/hip_bf16.h>

// ---------------------------------------------------------------------------
// LSTM (B=1024, T=100, D=512, H=256) + dense head, MI455X (gfx1250).
// bf16 WMMA 16x16x32 with f32 accumulate; all fp32->bf16 conversion hoisted
// out of the hot loop (x pre-packed once; h written packed by the step).
// Hot loop: fully unrolled, straight-line loads + WMMAs, no arrays/branches,
// __launch_bounds__(32,1) so the wave owns the register file (no spills).
// ---------------------------------------------------------------------------

#define B_   1024
#define T_   100
#define D_   512
#define H_   256
#define G4H  (4 * H_)
#define OUT_ 3

#define NKT_X (D_ / 32)        // 16 k-tiles for x@Wx
#define NKT_H (H_ / 32)        // 8 k-tiles for h@Wh
#define NRT   (B_ / 16)        // 64 row tiles
#define TILE_SHORTS 512        // 32 lanes * 16 shorts per (tile, ktile)

typedef __attribute__((ext_vector_type(16))) __bf16 v16bf;
typedef __attribute__((ext_vector_type(8)))  float  v8f;

__device__ __forceinline__ unsigned short f2bf(float f) {
    unsigned int u = __float_as_uint(f);
    unsigned int r = u + 0x7FFFu + ((u >> 16) & 1u);
    return (unsigned short)(r >> 16);
}

__device__ __forceinline__ float sigmoidf_(float x) {
    return 1.0f / (1.0f + __expf(-x));
}

__device__ __forceinline__ v16bf ld16(const unsigned short* __restrict__ p) {
    union { v16bf v; uint4 q[2]; } u;
    u.q[0] = *(const uint4*)(p);
    u.q[1] = *(const uint4*)(p + 8);
    return u.v;
}

// A/B-operand element mapping: lane L, elem e ->
// k = (e<8 ? (L>>4)*8 + e : 16 + (L>>4)*8 + (e-8))
__device__ __forceinline__ int a_elem_k(int lane, int e) {
    int base = (lane >> 4) * 8;
    return (e < 8) ? (base + e) : (16 + base + (e - 8));
}

#define WMMA_BF16(A, Bm, C) \
    __builtin_amdgcn_wmma_f32_16x16x32_bf16(false, (A), false, (Bm), (short)0, (C), false, false)

// ---------------------------------------------------------------------------
// Packing kernels (run once, HBM-bandwidth bound, negligible vs GEMM work)
// ---------------------------------------------------------------------------
__global__ void pack_x_kernel(const float* __restrict__ x,
                              unsigned short* __restrict__ dst) {
    long long i = (long long)blockIdx.x * blockDim.x + threadIdx.x;
    if (i >= (long long)B_ * T_ * D_) return;
    int e    = (int)(i & 15);
    int lane = (int)((i >> 4) & 31);
    int kt   = (int)((i >> 9) & 15);
    int rt   = (int)((i >> 13) & 63);
    int t    = (int)(i >> 19);
    int k   = kt * 32 + a_elem_k(lane, e);
    int row = rt * 16 + (lane & 15);
    dst[i] = f2bf(x[((size_t)row * T_ + t) * D_ + k]);
}

__global__ void pack_h0_kernel(const float* __restrict__ h0,
                               unsigned short* __restrict__ dst) {
    int i = blockIdx.x * blockDim.x + threadIdx.x;
    if (i >= B_ * H_) return;
    int e    = i & 15;
    int lane = (i >> 4) & 31;
    int kt   = (i >> 9) & 7;
    int rt   = i >> 12;
    int k   = kt * 32 + a_elem_k(lane, e);
    int row = rt * 16 + (lane & 15);
    dst[i] = f2bf(h0[(size_t)row * H_ + k]);
}

__global__ void pack_w_kernel(const float* __restrict__ W,
                              unsigned short* __restrict__ pk,
                              int K, int N, int nkt) {
    int i = blockIdx.x * blockDim.x + threadIdx.x;
    if (i >= K * N) return;
    int e     = i & 15;
    int lane  = (i >> 4) & 31;
    int tile  = i >> 9;
    int kt    = tile % nkt;
    int ntile = tile / nkt;
    int n = ntile * 16 + (lane & 15);
    int k = kt * 32 + a_elem_k(lane, e);
    pk[i] = f2bf(W[(size_t)k * N + n]);
}

__global__ void init_state_kernel(const float* __restrict__ h0,
                                  const float* __restrict__ c0,
                                  float* __restrict__ hdst,
                                  float* __restrict__ cdst, int n) {
    int i = blockIdx.x * blockDim.x + threadIdx.x;
    if (i >= n) return;
    hdst[i] = h0[i];
    cdst[i] = c0[i];
}

// ---------------------------------------------------------------------------
// Gate tail for one 16x16 accumulator group (rr-th row tile).
// Accumulator layout: lane -> column N = lane&15; component r -> row
// M = r + 8*(lane>>4).
// ---------------------------------------------------------------------------
__device__ __forceinline__ void gate_tail(
    v8f zi, v8f zf, v8f zg, v8f zo,
    int row0, int hi, int col,
    float bi, float bf, float bg, float bo,
    const float* __restrict__ cprev,
    float* __restrict__ hnext, float* __restrict__ cnext,
    unsigned short* __restrict__ hPkNext,
    int kt_h, int lanehi, int pe) {
#pragma unroll
    for (int r = 0; r < 8; ++r) {
        const int row = row0 + hi + r;
        float iv = sigmoidf_(zi[r] + bi);
        float fv = sigmoidf_(zf[r] + bf);
        float gv = tanhf    (zg[r] + bg);
        float ov = sigmoidf_(zo[r] + bo);
        float cold = cprev[(size_t)row * H_ + col];
        float cn = fv * cold + iv * gv;
        float hn = ov * tanhf(cn);
        cnext[(size_t)row * H_ + col] = cn;
        hnext[(size_t)row * H_ + col] = hn;
        const int m     = row & 15;
        const int rtw   = row >> 4;
        const int plane = m + 16 * lanehi;
        hPkNext[(((size_t)rtw * NKT_H + kt_h) * 32 + plane) * 16 + pe] = f2bf(hn);
    }
}

// ---------------------------------------------------------------------------
// One LSTM timestep.  grid = (B/32, H/16), block = 32 (one wave).
// Wave: 2 row-tiles x 4 gates = 8 f32 accumulators (64 VGPRs).
// Per k-step: 6 x 32B tile loads -> 8 WMMAs (128 KFLOP / 192 B).
// ---------------------------------------------------------------------------
__global__ void __launch_bounds__(32, 1)
lstm_step_kernel(const unsigned short* __restrict__ xPk,
                 const unsigned short* __restrict__ WxP,
                 const unsigned short* __restrict__ WhP,
                 const float* __restrict__ bias,
                 const unsigned short* __restrict__ hPkPrev,
                 const float* __restrict__ cprev,
                 float* __restrict__ hnext,
                 float* __restrict__ cnext,
                 unsigned short* __restrict__ hPkNext,
                 int t) {
    const int lane = threadIdx.x;
    const int rt0  = blockIdx.x * 2;   // first of two 16-row tiles
    const int ct   = blockIdx.y;       // h-column tile (0..15)

    v8f acc00 = (v8f){0,0,0,0,0,0,0,0}, acc01 = acc00, acc02 = acc00, acc03 = acc00;
    v8f acc10 = acc00, acc11 = acc00, acc12 = acc00, acc13 = acc00;

    // ---------------- phase 1: x_t @ Wx (16 k-steps, fully unrolled) --------
    {
        const unsigned short* aP0 =
            xPk + (((size_t)t * NRT + rt0) * NKT_X) * TILE_SHORTS + lane * 16;
        const unsigned short* aP1 = aP0 + (size_t)NKT_X * TILE_SHORTS;
        const unsigned short* bP0 =
            WxP + ((size_t)(0 * 16 + ct) * NKT_X) * TILE_SHORTS + lane * 16;
        const unsigned short* bP1 =
            WxP + ((size_t)(1 * 16 + ct) * NKT_X) * TILE_SHORTS + lane * 16;
        const unsigned short* bP2 =
            WxP + ((size_t)(2 * 16 + ct) * NKT_X) * TILE_SHORTS + lane * 16;
        const unsigned short* bP3 =
            WxP + ((size_t)(3 * 16 + ct) * NKT_X) * TILE_SHORTS + lane * 16;

#pragma unroll
        for (int kt = 0; kt < NKT_X; ++kt) {
            const size_t off = (size_t)kt * TILE_SHORTS;
            v16bf a0 = ld16(aP0 + off);
            v16bf a1 = ld16(aP1 + off);
            v16bf b0 = ld16(bP0 + off);
            v16bf b1 = ld16(bP1 + off);
            v16bf b2 = ld16(bP2 + off);
            v16bf b3 = ld16(bP3 + off);
            acc00 = WMMA_BF16(a0, b0, acc00);
            acc10 = WMMA_BF16(a1, b0, acc10);
            acc01 = WMMA_BF16(a0, b1, acc01);
            acc11 = WMMA_BF16(a1, b1, acc11);
            acc02 = WMMA_BF16(a0, b2, acc02);
            acc12 = WMMA_BF16(a1, b2, acc12);
            acc03 = WMMA_BF16(a0, b3, acc03);
            acc13 = WMMA_BF16(a1, b3, acc13);
        }
    }

    // ---------------- phase 2: h_{t-1} @ Wh (8 k-steps, fully unrolled) -----
    {
        const unsigned short* aP0 =
            hPkPrev + ((size_t)rt0 * NKT_H) * TILE_SHORTS + lane * 16;
        const unsigned short* aP1 = aP0 + (size_t)NKT_H * TILE_SHORTS;
        const unsigned short* bP0 =
            WhP + ((size_t)(0 * 16 + ct) * NKT_H) * TILE_SHORTS + lane * 16;
        const unsigned short* bP1 =
            WhP + ((size_t)(1 * 16 + ct) * NKT_H) * TILE_SHORTS + lane * 16;
        const unsigned short* bP2 =
            WhP + ((size_t)(2 * 16 + ct) * NKT_H) * TILE_SHORTS + lane * 16;
        const unsigned short* bP3 =
            WhP + ((size_t)(3 * 16 + ct) * NKT_H) * TILE_SHORTS + lane * 16;

#pragma unroll
        for (int kt = 0; kt < NKT_H; ++kt) {
            const size_t off = (size_t)kt * TILE_SHORTS;
            v16bf a0 = ld16(aP0 + off);
            v16bf a1 = ld16(aP1 + off);
            v16bf b0 = ld16(bP0 + off);
            v16bf b1 = ld16(bP1 + off);
            v16bf b2 = ld16(bP2 + off);
            v16bf b3 = ld16(bP3 + off);
            acc00 = WMMA_BF16(a0, b0, acc00);
            acc10 = WMMA_BF16(a1, b0, acc10);
            acc01 = WMMA_BF16(a0, b1, acc01);
            acc11 = WMMA_BF16(a1, b1, acc11);
            acc02 = WMMA_BF16(a0, b2, acc02);
            acc12 = WMMA_BF16(a1, b2, acc12);
            acc03 = WMMA_BF16(a0, b3, acc03);
            acc13 = WMMA_BF16(a1, b3, acc13);
        }
    }

    // ---------------- gates + state update ----------------
    const int col = ct * 16 + (lane & 15);
    const float bi = bias[0 * H_ + col];
    const float bf = bias[1 * H_ + col];
    const float bg = bias[2 * H_ + col];
    const float bo = bias[3 * H_ + col];
    const int hi = (lane >> 4) * 8;

    // packed-h scatter coordinates (this lane's h column = k of next GEMM)
    const int kt_h   = col >> 5;
    const int kk     = col & 31;
    const int seg    = kk >> 4;
    const int sub    = kk & 15;
    const int lanehi = (sub >= 8) ? 1 : 0;
    const int pe     = (seg == 0) ? (sub - 8 * lanehi) : (8 + sub - 8 * lanehi);

    gate_tail(acc00, acc01, acc02, acc03, rt0 * 16, hi, col,
              bi, bf, bg, bo, cprev, hnext, cnext, hPkNext, kt_h, lanehi, pe);
    gate_tail(acc10, acc11, acc12, acc13, (rt0 + 1) * 16, hi, col,
              bi, bf, bg, bo, cprev, hnext, cnext, hPkNext, kt_h, lanehi, pe);
}

// ---------------------------------------------------------------------------
// Dense head: out[b,o] = h[b,:] . Wd[:,o] + bd[o]   (1024x256 @ 256x3, tiny)
// ---------------------------------------------------------------------------
__global__ void head_kernel(const float* __restrict__ h,
                            const float* __restrict__ Wd,
                            const float* __restrict__ bd,
                            float* __restrict__ out) {
    int i = blockIdx.x * blockDim.x + threadIdx.x;
    if (i >= B_ * OUT_) return;
    int b = i / OUT_;
    int o = i % OUT_;
    float s = bd[o];
#pragma unroll 8
    for (int j = 0; j < H_; ++j)
        s += h[(size_t)b * H_ + j] * Wd[(size_t)j * OUT_ + o];
    out[i] = s;
}

extern "C" void kernel_launch(void* const* d_in, const int* in_sizes, int n_in,
                              void* d_out, int out_size, void* d_ws, size_t ws_size,
                              hipStream_t stream) {
    (void)in_sizes; (void)n_in; (void)out_size; (void)ws_size;

    const float* x  = (const float*)d_in[0];
    const float* h0 = (const float*)d_in[1];
    const float* c0 = (const float*)d_in[2];
    const float* Wx = (const float*)d_in[3];
    const float* Wh = (const float*)d_in[4];
    const float* bv = (const float*)d_in[5];
    const float* Wd = (const float*)d_in[6];
    const float* bd = (const float*)d_in[7];
    float* out = (float*)d_out;

    // ---- workspace layout (~112 MB) ----
    char* ws = (char*)d_ws;
    const size_t MB = 1024 * 1024;
    float* hbuf[2] = { (float*)(ws + 0 * MB), (float*)(ws + 1 * MB) };
    float* cbuf[2] = { (float*)(ws + 2 * MB), (float*)(ws + 3 * MB) };
    unsigned short* WxP  = (unsigned short*)(ws + 4 * MB);
    unsigned short* WhP  = (unsigned short*)(ws + 5 * MB);
    unsigned short* hPk0 = (unsigned short*)(ws + 5 * MB + 512 * 1024);
    unsigned short* hPk1 = (unsigned short*)(ws + 6 * MB);
    unsigned short* xPk  = (unsigned short*)(ws + 6 * MB + 512 * 1024);
    unsigned short* hPk[2] = { hPk0, hPk1 };

    // 1) pre-pack x, weights, h0 into bf16 operand tiles
    {
        long long n = (long long)B_ * T_ * D_;
        pack_x_kernel<<<(unsigned)((n + 255) / 256), 256, 0, stream>>>(x, xPk);
        int nw = D_ * G4H;
        pack_w_kernel<<<(nw + 255) / 256, 256, 0, stream>>>(Wx, WxP, D_, G4H, NKT_X);
        nw = H_ * G4H;
        pack_w_kernel<<<(nw + 255) / 256, 256, 0, stream>>>(Wh, WhP, H_, G4H, NKT_H);
        int nh = B_ * H_;
        pack_h0_kernel<<<(nh + 255) / 256, 256, 0, stream>>>(h0, hPk0);
        init_state_kernel<<<(nh + 255) / 256, 256, 0, stream>>>(h0, c0, hbuf[0], cbuf[0], nh);
    }

    // 2) 100 sequential timesteps (graph-captured launches)
    dim3 grid(B_ / 32, H_ / 16);
    for (int t = 0; t < T_; ++t) {
        int cur = t & 1, nxt = (t + 1) & 1;
        lstm_step_kernel<<<grid, 32, 0, stream>>>(
            xPk, WxP, WhP, bv, hPk[cur], cbuf[cur],
            hbuf[nxt], cbuf[nxt], hPk[nxt], t);
    }

    // 3) dense head on h_T (fp32 copy lands in buffer T_&1 == 0)
    {
        int n = B_ * OUT_;
        head_kernel<<<(n + 255) / 256, 256, 0, stream>>>(hbuf[T_ & 1], Wd, bd, out);
    }
}
